// GATv2_12017318494741
// MI455X (gfx1250) — matrix operand order repaired and checked by reading, not compiled, not run on hardware
//
#include <hip/hip_runtime.h>

typedef __attribute__((ext_vector_type(16))) __bf16 bf16x16;
typedef __attribute__((ext_vector_type(8)))  __bf16 bf16x8;
typedef __attribute__((ext_vector_type(8)))  float  f32x8;

static constexpr int   kN     = 50000;
static constexpr int   kE     = 800000;
static constexpr int   kEP    = kE + kN;   // edges + self-loops
static constexpr int   kIN    = 128;
static constexpr int   kHC    = 256;       // H * C
static constexpr int   kH     = 8;
static constexpr int   kC     = 32;
static constexpr float kEPSBN = 1e-5f;
static constexpr float kSLOPE = 0.2f;

// ---------------- helpers ----------------

__device__ __forceinline__ unsigned short f32_to_bf16_rne(float f) {
  unsigned int u = __float_as_uint(f);
  u += 0x7fffu + ((u >> 16) & 1u);
  return (unsigned short)(u >> 16);
}

__device__ __forceinline__ bf16x16 ld_frag(const unsigned short* p0,
                                           const unsigned short* p1) {
  bf16x8 lo = *reinterpret_cast<const bf16x8*>(p0);
  bf16x8 hi = *reinterpret_cast<const bf16x8*>(p1);
  return __builtin_shufflevector(lo, hi, 0,1,2,3,4,5,6,7,8,9,10,11,12,13,14,15);
}

__device__ __forceinline__ void atomicMaxF(float* addr, float value) {
  if (value >= 0.f) atomicMax((int*)addr, __float_as_int(value));
  else              atomicMin((unsigned int*)addr, (unsigned int)__float_as_int(value));
}

// ---------------- conversion / init kernels ----------------

__global__ void gatv2_cvt_bf16(const float* __restrict__ in,
                               unsigned short* __restrict__ out, int n) {
  int i = blockIdx.x * blockDim.x + threadIdx.x;
  if (i < n) out[i] = f32_to_bf16_rne(in[i]);
}

// Pack Wl (K x 256) and Wr (K x 256) fp32 into one transposed bf16 matrix
// Wt [512 x K]: rows 0..255 = Wl columns, rows 256..511 = Wr columns.
__global__ void gatv2_pack_w(const float* __restrict__ Wl,
                             const float* __restrict__ Wr,
                             unsigned short* __restrict__ Wt, int K) {
  int i = blockIdx.x * blockDim.x + threadIdx.x;
  if (i >= 512 * K) return;
  int n = i / K, k = i - n * K;
  float v = (n < kHC) ? Wl[(size_t)k * kHC + n] : Wr[(size_t)k * kHC + (n - kHC)];
  Wt[i] = f32_to_bf16_rne(v);
}

__global__ void gatv2_fill(float* __restrict__ p, float v, long long n) {
  long long i = (long long)blockIdx.x * blockDim.x + threadIdx.x;
  if (i < n) p[i] = v;
}

// ---------------- WMMA GEMM: C[M x Nc] = A[M x K] * Bt[Nc x K]^T ----------------
// One wave (32 threads) per 16x64 output strip; bf16 inputs, f32 accumulate.
__global__ void gatv2_wmma_gemm(const unsigned short* __restrict__ A,
                                const unsigned short* __restrict__ Bt,
                                float* __restrict__ C, int M, int K, int Nc) {
  const int lane = threadIdx.x & 31;
  const int m0   = blockIdx.x << 4;
  const int n0   = blockIdx.y << 6;
  const int s    = lane >> 4;        // half-wave selector
  const int r16  = lane & 15;

  const unsigned short* arow = A  + (size_t)(m0 + r16) * K;
  const unsigned short* b0r  = Bt + (size_t)(n0 +  0 + r16) * K;
  const unsigned short* b1r  = Bt + (size_t)(n0 + 16 + r16) * K;
  const unsigned short* b2r  = Bt + (size_t)(n0 + 32 + r16) * K;
  const unsigned short* b3r  = Bt + (size_t)(n0 + 48 + r16) * K;

  f32x8 acc0 = {0.f,0.f,0.f,0.f,0.f,0.f,0.f,0.f};
  f32x8 acc1 = acc0, acc2 = acc0, acc3 = acc0;

  for (int k0 = 0; k0 < K; k0 += 32) {
    const int ao = k0 + s * 16;   // A: lanes<16 -> K 0..15; lanes>=16 -> K 16..31
    const int bo = k0 + s * 16;   // B: lanes<16 -> K 0..15; lanes>=16 -> K 16..31
    bf16x16 a  = ld_frag(arow + ao, arow + ao + 8);
    bf16x16 b0 = ld_frag(b0r + bo, b0r + bo + 8);
    acc0 = __builtin_amdgcn_wmma_f32_16x16x32_bf16(false, a, false, b0, (short)0, acc0, false, false);
    bf16x16 b1 = ld_frag(b1r + bo, b1r + bo + 8);
    acc1 = __builtin_amdgcn_wmma_f32_16x16x32_bf16(false, a, false, b1, (short)0, acc1, false, false);
    bf16x16 b2 = ld_frag(b2r + bo, b2r + bo + 8);
    acc2 = __builtin_amdgcn_wmma_f32_16x16x32_bf16(false, a, false, b2, (short)0, acc2, false, false);
    bf16x16 b3 = ld_frag(b3r + bo, b3r + bo + 8);
    acc3 = __builtin_amdgcn_wmma_f32_16x16x32_bf16(false, a, false, b3, (short)0, acc3, false, false);
  }

  // C/D layout: VGPR r -> row r (lanes 0-15) / row 8+r (lanes 16-31), col = lane&15
  #pragma unroll
  for (int r = 0; r < 8; ++r) {
    float* cp = C + (size_t)(m0 + s * 8 + r) * Nc + n0 + r16;
    cp[0]  = acc0[r];
    cp[16] = acc1[r];
    cp[32] = acc2[r];
    cp[48] = acc3[r];
  }
}

// ---------------- edge kernels (wave per edge; lane owns 8 contiguous channels) --------

__device__ __forceinline__ void edge_endpoints(const long long* ei, int w, int& src, int& dst) {
  if (w < kE) { src = (int)ei[w]; dst = (int)ei[kE + w]; }
  else        { src = w - kE; dst = src; }           // self-loop
}

// logits[e][h] = att[h] . leakyrelu(xl[src] + xr[dst]); also atomic segment-max over dst
__global__ void gatv2_edge_logits(const long long* __restrict__ ei,
                                  const float* __restrict__ xlr,  // [N x 512] = [xl | xr]
                                  const float* __restrict__ att,  // [256]
                                  float* __restrict__ elog,       // [EP x 8]
                                  float* __restrict__ nmax) {     // [N x 8]
  int gid = blockIdx.x * blockDim.x + threadIdx.x;
  int w = gid >> 5, lane = gid & 31;
  if (w >= kEP) return;
  int src, dst; edge_endpoints(ei, w, src, dst);
  const float* xl = xlr + (size_t)src * 512;
  const float* xr = xlr + (size_t)dst * 512 + kHC;
  const int base = lane * 8;
  float sacc = 0.f;
  #pragma unroll
  for (int u = 0; u < 8; ++u) {
    int c = base + u;
    float e = xl[c] + xr[c];
    e = (e > 0.f) ? e : kSLOPE * e;
    sacc = fmaf(e, att[c], sacc);
  }
  // reduce the 4 lanes that cover one head (32 channels)
  sacc += __shfl_xor(sacc, 1);
  sacc += __shfl_xor(sacc, 2);
  if ((lane & 3) == 0) {
    int h = lane >> 2;
    elog[(size_t)w * kH + h] = sacc;
    atomicMaxF(&nmax[(size_t)dst * kH + h], sacc);
  }
}

// ex = exp(logit - max[dst]); denom[dst] += ex; thread per (edge, head)
__global__ void gatv2_edge_exp(const long long* __restrict__ ei,
                               const float* __restrict__ nmax,
                               float* __restrict__ elog,
                               float* __restrict__ nden) {
  int i = blockIdx.x * blockDim.x + threadIdx.x;
  if (i >= kEP * kH) return;
  int w = i >> 3, h = i & 7;
  int dst = (w < kE) ? (int)ei[kE + w] : (w - kE);
  float ex = __expf(elog[i] - nmax[(size_t)dst * kH + h]);
  elog[i] = ex;
  atomicAdd(&nden[(size_t)dst * kH + h], ex);
}

// out[dst] += (ex/denom[dst]) * xl[src]; wave per edge
__global__ void gatv2_edge_aggregate(const long long* __restrict__ ei,
                                     const float* __restrict__ xlr,
                                     const float* __restrict__ elog,
                                     const float* __restrict__ nden,
                                     float* __restrict__ aggr) {  // [N x 256]
  int gid = blockIdx.x * blockDim.x + threadIdx.x;
  int w = gid >> 5, lane = gid & 31;
  if (w >= kEP) return;
  int src, dst; edge_endpoints(ei, w, src, dst);
  int h = lane >> 2;
  float alpha = elog[(size_t)w * kH + h] / nden[(size_t)dst * kH + h];
  const float* xl = xlr + (size_t)src * 512 + lane * 8;
  float*       op = aggr + (size_t)dst * kHC + lane * 8;
  #pragma unroll
  for (int u = 0; u < 8; ++u) atomicAdd(&op[u], alpha * xl[u]);
}

// ---------------- per-node epilogues ----------------

// layer0: +b0, batchnorm, ELU, cast to bf16 for the layer-1 GEMM
__global__ void gatv2_post0(const float* __restrict__ aggr,
                            const float* __restrict__ b0, const float* __restrict__ g0,
                            const float* __restrict__ be0, const float* __restrict__ m0,
                            const float* __restrict__ v0,
                            unsigned short* __restrict__ hbf) {
  int i = blockIdx.x * blockDim.x + threadIdx.x;
  if (i >= kN * kHC) return;
  int c = i & (kHC - 1);
  float x = aggr[i] + b0[c];
  x = (x - m0[c]) * rsqrtf(v0[c] + kEPSBN) * g0[c] + be0[c];
  x = (x > 0.f) ? x : (__expf(x) - 1.f);   // ELU
  hbf[i] = f32_to_bf16_rne(x);
}

// layer1: mean over heads, +b1, batchnorm, classifier (32 -> 2). Wave per node.
__global__ void gatv2_post1(const float* __restrict__ aggr,
                            const float* __restrict__ b1, const float* __restrict__ g1,
                            const float* __restrict__ be1, const float* __restrict__ m1,
                            const float* __restrict__ v1,
                            const float* __restrict__ Wc, const float* __restrict__ bc,
                            float* __restrict__ out) {
  int gid = blockIdx.x * blockDim.x + threadIdx.x;
  int node = gid >> 5, lane = gid & 31;
  if (node >= kN) return;
  const float* a = aggr + (size_t)node * kHC;
  float s = 0.f;
  #pragma unroll
  for (int h = 0; h < kH; ++h) s += a[h * kC + lane];
  s = s * (1.f / kH) + b1[lane];
  s = (s - m1[lane]) * rsqrtf(v1[lane] + kEPSBN) * g1[lane] + be1[lane];
  float p0 = s * Wc[lane * 2 + 0];
  float p1 = s * Wc[lane * 2 + 1];
  #pragma unroll
  for (int o = 1; o < 32; o <<= 1) { p0 += __shfl_xor(p0, o); p1 += __shfl_xor(p1, o); }
  if (lane == 0) {
    out[(size_t)node * 2 + 0] = p0 + bc[0];
    out[(size_t)node * 2 + 1] = p1 + bc[1];
  }
}

// ---------------- launch ----------------

static inline size_t align256(size_t x) { return (x + 255) & ~(size_t)255; }

extern "C" void kernel_launch(void* const* d_in, const int* in_sizes, int n_in,
                              void* d_out, int out_size, void* d_ws, size_t ws_size,
                              hipStream_t stream) {
  const float*     x    = (const float*)d_in[0];
  const long long* ei   = (const long long*)d_in[1];   // int64 per reference
  const float*     Wl0  = (const float*)d_in[2];
  const float*     Wr0  = (const float*)d_in[3];
  const float*     att0 = (const float*)d_in[4];
  const float*     b0   = (const float*)d_in[5];
  const float*     g0   = (const float*)d_in[6];
  const float*     be0  = (const float*)d_in[7];
  const float*     m0   = (const float*)d_in[8];
  const float*     v0   = (const float*)d_in[9];
  const float*     Wl1  = (const float*)d_in[10];
  const float*     Wr1  = (const float*)d_in[11];
  const float*     att1 = (const float*)d_in[12];
  const float*     b1   = (const float*)d_in[13];
  const float*     g1   = (const float*)d_in[14];
  const float*     be1  = (const float*)d_in[15];
  const float*     m1   = (const float*)d_in[16];
  const float*     v1   = (const float*)d_in[17];
  const float*     Wc   = (const float*)d_in[18];
  const float*     bc   = (const float*)d_in[19];
  float*           out  = (float*)d_out;

  char* ws = (char*)d_ws;
  size_t off = 0;
  unsigned short* xbf = (unsigned short*)(ws + off); off += align256((size_t)kN * kIN * 2);
  unsigned short* w0t = (unsigned short*)(ws + off); off += align256((size_t)512 * kIN * 2);
  unsigned short* w1t = (unsigned short*)(ws + off); off += align256((size_t)512 * kHC * 2);
  unsigned short* hbf = (unsigned short*)(ws + off); off += align256((size_t)kN * kHC * 2);
  float* xlr  = (float*)(ws + off); off += align256((size_t)kN * 512 * 4);   // reused both layers
  float* elog = (float*)(ws + off); off += align256((size_t)kEP * kH * 4);
  float* nmax = (float*)(ws + off); off += align256((size_t)kN * kH * 4);
  float* nden = (float*)(ws + off); off += align256((size_t)kN * kH * 4);
  float* aggr = (float*)(ws + off); off += align256((size_t)kN * kHC * 4);   // reused both layers

  const int TB = 256;
  const float NEG_INF = -__builtin_huge_valf();
  const int gEdgeWave = (kEP * 32 + TB - 1) / TB;   // wave-per-edge kernels
  const int gEdgeEl   = (kEP * kH + TB - 1) / TB;
  const int gNodeEl   = (kN * kHC + TB - 1) / TB;
  const int gNH       = (kN * kH + TB - 1) / TB;

  // ---- prep: bf16 inputs + fused transposed weights ----
  gatv2_cvt_bf16<<<(kN * kIN + TB - 1) / TB, TB, 0, stream>>>(x, xbf, kN * kIN);
  gatv2_pack_w<<<(512 * kIN + TB - 1) / TB, TB, 0, stream>>>(Wl0, Wr0, w0t, kIN);
  gatv2_pack_w<<<(512 * kHC + TB - 1) / TB, TB, 0, stream>>>(Wl1, Wr1, w1t, kHC);

  // ---- layer 0 ----
  gatv2_wmma_gemm<<<dim3(kN / 16, 512 / 64), 32, 0, stream>>>(xbf, w0t, xlr, kN, kIN, 512);
  gatv2_fill<<<gNH, TB, 0, stream>>>(nmax, NEG_INF, (long long)kN * kH);
  gatv2_fill<<<gNH, TB, 0, stream>>>(nden, 0.f, (long long)kN * kH);
  gatv2_fill<<<gNodeEl, TB, 0, stream>>>(aggr, 0.f, (long long)kN * kHC);
  gatv2_edge_logits<<<gEdgeWave, TB, 0, stream>>>(ei, xlr, att0, elog, nmax);
  gatv2_edge_exp<<<gEdgeEl, TB, 0, stream>>>(ei, nmax, elog, nden);
  gatv2_edge_aggregate<<<gEdgeWave, TB, 0, stream>>>(ei, xlr, elog, nden, aggr);
  gatv2_post0<<<gNodeEl, TB, 0, stream>>>(aggr, b0, g0, be0, m0, v0, hbf);

  // ---- layer 1 ----
  gatv2_wmma_gemm<<<dim3(kN / 16, 512 / 64), 32, 0, stream>>>(hbf, w1t, xlr, kN, kHC, 512);
  gatv2_fill<<<gNH, TB, 0, stream>>>(nmax, NEG_INF, (long long)kN * kH);
  gatv2_fill<<<gNH, TB, 0, stream>>>(nden, 0.f, (long long)kN * kH);
  gatv2_fill<<<gNodeEl, TB, 0, stream>>>(aggr, 0.f, (long long)kN * kHC);
  gatv2_edge_logits<<<gEdgeWave, TB, 0, stream>>>(ei, xlr, att1, elog, nmax);
  gatv2_edge_exp<<<gEdgeEl, TB, 0, stream>>>(ei, nmax, elog, nden);
  gatv2_edge_aggregate<<<gEdgeWave, TB, 0, stream>>>(ei, xlr, elog, nden, aggr);
  gatv2_post1<<<(kN * 32 + TB - 1) / TB, TB, 0, stream>>>(aggr, b1, g1, be1, m1, v1, Wc, bc, out);
}
